// GenBNN_71433896067212
// MI455X (gfx1250) — 8-bit matrix operand order repaired and checked by reading, compiled, not run on hardware
//
#include <hip/hip_runtime.h>
#include <hip/hip_bf16.h>
#include <math.h>

typedef __attribute__((ext_vector_type(8))) int v8i_t;

#define BATCH   65536
#define HID     512
#define KDIM    512

__global__ __launch_bounds__(256) void bnn_sign_f32(
    const float* __restrict__ src, signed char* __restrict__ dst, long long n)
{
    long long i = (long long)blockIdx.x * blockDim.x + threadIdx.x;
    if (i >= n) return;
    float v = src[i];
    dst[i] = (v > 0.f) ? (signed char)1 : ((v < 0.f) ? (signed char)-1 : (signed char)0);
}

__global__ __launch_bounds__(256) void bnn_gemm_iu8(
    const signed char* __restrict__ A,
    const signed char* __restrict__ B,
    short* __restrict__ C,
    unsigned long long* __restrict__ sum,
    unsigned long long* __restrict__ sumsq,
    int M, int N, int do_stats)
{
    constexpr int K = KDIM;
    const int wave = threadIdx.x >> 5;
    const int lane = threadIdx.x & 31;
    const int tiles_n = N >> 4;
    const int tile = blockIdx.x * 8 + wave;
    const int tm = tile / tiles_n;
    const int tn = tile - tm * tiles_n;
    if (tm * 16 >= M) return;

    const int hi = lane >> 4;
    const int l  = lane & 15;

    const signed char* arow = A + (size_t)(tm * 16 + l) * K;
    const signed char* brow = B + (size_t)(tn * 16 + l) * K;

    v8i_t acc = {};
#pragma unroll
    for (int k0 = 0; k0 < K; k0 += 64) {
        v8i_t av, bv;
#pragma unroll
        for (int j = 0; j < 4; ++j) {
            int2 d = *(const int2*)(arow + k0 + (j >> 1) * 32 + hi * 16 + (j & 1) * 8);
            av[2 * j]     = d.x;
            av[2 * j + 1] = d.y;
        }
        int4 b0 = *(const int4*)(brow + k0 + hi * 16);
        int4 b1 = *(const int4*)(brow + k0 + 32 + hi * 16);
        bv[0] = b0.x; bv[1] = b0.y; bv[2] = b0.z; bv[3] = b0.w;
        bv[4] = b1.x; bv[5] = b1.y; bv[6] = b1.z; bv[7] = b1.w;

        acc = __builtin_amdgcn_wmma_i32_16x16x64_iu8(
             true, av,  true, bv, acc,
             false,  false);
    }

    short* crow = C + (size_t)(tm * 16 + hi * 8) * N + tn * 16 + l;
#pragma unroll
    for (int r = 0; r < 8; ++r)
        crow[(size_t)r * N] = (short)acc[r];

    if (do_stats) {
        int ps = 0, qs = 0;
#pragma unroll
        for (int r = 0; r < 8; ++r) { ps += acc[r]; qs += acc[r] * acc[r]; }
        ps += __shfl_xor(ps, 16, 32);
        qs += __shfl_xor(qs, 16, 32);
        if (lane < 16) {
            atomicAdd(&sum[tn * 16 + l],   (unsigned long long)(long long)ps);
            atomicAdd(&sumsq[tn * 16 + l], (unsigned long long)(long long)qs);
        }
    }
}

__global__ __launch_bounds__(256) void bnn_bnfinalize(
    const unsigned long long* __restrict__ sum,
    const unsigned long long* __restrict__ sumsq,
    const float* __restrict__ gamma, const float* __restrict__ beta,
    float* __restrict__ scale, float* __restrict__ shift, int M, int N)
{
    int c = blockIdx.x * blockDim.x + threadIdx.x;
    if (c >= N) return;
    double s = (double)(long long)sum[c];
    double q = (double)(long long)sumsq[c];
    double mean = s / (double)M;
    double var  = q / (double)M - mean * mean;
    float rs = rsqrtf((float)var + 1e-5f);
    float sc = gamma[c] * rs;
    scale[c] = sc;
    shift[c] = beta[c] - (float)mean * sc;
}

__global__ __launch_bounds__(256) void bnn_bnsign(
    const short* __restrict__ C, const float* __restrict__ scale,
    const float* __restrict__ shift, signed char* __restrict__ Anext,
    long long total)
{
    long long idx = (long long)blockIdx.x * blockDim.x + threadIdx.x;
    if (idx >= total) return;
    int c = (int)(idx & (HID - 1));
    float z = (float)C[idx] * scale[c] + shift[c];
    Anext[idx] = (z > 0.f) ? (signed char)1 : ((z < 0.f) ? (signed char)-1 : (signed char)0);
}

__global__ __launch_bounds__(256) void bnn_logsoftmax(
    const short* __restrict__ C, const float* __restrict__ b_out,
    float* __restrict__ out, int M)
{
    int b = blockIdx.x * blockDim.x + threadIdx.x;
    if (b >= M) return;
    const short* crow = C + (size_t)b * 16;
    float v[10];
    float mx = -INFINITY;
#pragma unroll
    for (int o = 0; o < 10; ++o) {
        v[o] = (float)crow[o] + b_out[o];
        mx = fmaxf(mx, v[o]);
    }
    float se = 0.f;
#pragma unroll
    for (int o = 0; o < 10; ++o) se += expf(v[o] - mx);
    float lse = mx + logf(se);
    float* orow = out + (size_t)b * 10;
#pragma unroll
    for (int o = 0; o < 10; ++o) orow[o] = v[o] - lse;
}

extern "C" void kernel_launch(void* const* d_in, const int* in_sizes, int n_in,
                              void* d_out, int out_size, void* d_ws, size_t ws_size,
                              hipStream_t stream) {
    const float* x      = (const float*)d_in[0];
    const float* Ws[3]  = { (const float*)d_in[1], (const float*)d_in[2], (const float*)d_in[3] };
    const float* gam[3] = { (const float*)d_in[7], (const float*)d_in[8], (const float*)d_in[9] };
    const float* bet[3] = { (const float*)d_in[10], (const float*)d_in[11], (const float*)d_in[12] };
    const float* W_out  = (const float*)d_in[13];
    const float* b_out  = (const float*)d_in[14];

    char* ws = (char*)d_ws;
    const size_t MB = 1ull << 20;
    signed char* A0    = (signed char*)(ws + 0);
    signed char* A1    = (signed char*)(ws + 32 * MB);
    short*       C     = (short*)(ws + 64 * MB);
    signed char* Wsign = (signed char*)(ws + 128 * MB);
    unsigned long long* sum   = (unsigned long long*)(ws + 129 * MB);
    unsigned long long* sumsq = sum + HID;
    float* scale = (float*)(ws + 129 * MB + 8192);
    float* shift = scale + HID;

    const int M = BATCH, N = HID;
    const long long MK = (long long)M * KDIM;

    bnn_sign_f32<<<(unsigned)(MK / 256), 256, 0, stream>>>(x, A0, MK);
    for (int l = 0; l < 3; ++l)
        bnn_sign_f32<<<(HID * KDIM) / 256, 256, 0, stream>>>(
            Ws[l], Wsign + (size_t)l * HID * KDIM, (long long)HID * KDIM);
    signed char* WoutS = Wsign + 3ull * HID * KDIM;
    hipMemsetAsync(WoutS, 0, 16 * KDIM, stream);
    bnn_sign_f32<<<(10 * KDIM + 255) / 256, 256, 0, stream>>>(W_out, WoutS, 10 * KDIM);

    signed char* Acur = A0, *Anext = A1;
    const int hid_tiles  = (M / 16) * (N / 16);
    const int hid_blocks = hid_tiles / 8;
    for (int l = 0; l < 3; ++l) {
        hipMemsetAsync(sum, 0, 2 * HID * sizeof(unsigned long long), stream);
        bnn_gemm_iu8<<<hid_blocks, 256, 0, stream>>>(
            Acur, Wsign + (size_t)l * HID * KDIM, C, sum, sumsq, M, N, 1);
        bnn_bnfinalize<<<(HID + 255) / 256, 256, 0, stream>>>(
            sum, sumsq, gam[l], bet[l], scale, shift, M, N);
        bnn_bnsign<<<(unsigned)((long long)M * N / 256), 256, 0, stream>>>(
            C, scale, shift, Anext, (long long)M * N);
        signed char* t = Acur; Acur = Anext; Anext = t;
    }

    const int out_blocks = (M / 16) / 8;
    bnn_gemm_iu8<<<out_blocks, 256, 0, stream>>>(
        Acur, WoutS, C, nullptr, nullptr, M, 16, 0);
    bnn_logsoftmax<<<(M + 255) / 256, 256, 0, stream>>>(C, b_out, (float*)d_out, M);
}
